// MSDeformMatchV2Attn_10136122818977
// MI455X (gfx1250) — matrix operand order repaired and checked by reading, not compiled, not run on hardware
//
#include <hip/hip_runtime.h>
#include <hip/hip_bf16.h>

typedef __attribute__((ext_vector_type(16))) __bf16 v16bf;
typedef __attribute__((ext_vector_type(8)))  float  v8f;
typedef __attribute__((ext_vector_type(4)))  int    v4i;
typedef unsigned long long u64;
typedef unsigned int u32;
typedef unsigned short u16;

#define LQ   2048
#define DMODEL 256
#define LIN  16384
#define NLVL 4
#define HWL  4096   // 64*64 per level
#define GRID_H 64
#define CLIP_MAX 3969  // (h-1)*(w-1)

#if __has_builtin(__builtin_amdgcn_global_load_async_to_lds_b128) && \
    __has_builtin(__builtin_amdgcn_s_wait_asynccnt)
#define HAVE_ASYNC_LDS 1
#else
#define HAVE_ASYNC_LDS 0
#endif

union BF16x16 {
    v16bf v;
    uint4 u[2];
};

__device__ __forceinline__ u16 f2bf(float f) {
    u32 u = __float_as_uint(f);
    u32 r = (u + 0x7FFFu + ((u >> 16) & 1u)) >> 16;   // RNE
    return (u16)r;
}

__device__ __forceinline__ u32 orderable(float f) {
    u32 u = __float_as_uint(f);
    return (u & 0x80000000u) ? ~u : (u | 0x80000000u);
}

// Insert candidate into sorted-descending top4 (u32 packed keys).
// Becomes a v_max_u32/v_min_u32 compare-swap chain (branchless).
__device__ __forceinline__ void ins4(u32 k, u32& t0, u32& t1, u32& t2, u32& t3) {
    t3 = max(t3, k);                                  // evict min(old t3, k)
    u32 h2 = max(t2, t3), l2 = min(t2, t3); t2 = h2; t3 = l2;
    u32 h1 = max(t1, t2), l1 = min(t1, t2); t1 = h1; t2 = l1;
    u32 h0 = max(t0, t1), l0 = min(t0, t1); t0 = h0; t1 = l0;
}

// Stage one 16-byte chunk global->LDS (async path when available).
__device__ __forceinline__ void stage16(const u16* gsrc, u16* ldst) {
#if HAVE_ASYNC_LDS
    __builtin_amdgcn_global_load_async_to_lds_b128(
        (__attribute__((address_space(1))) v4i*)(uintptr_t)gsrc,
        (__attribute__((address_space(3))) v4i*)(uintptr_t)ldst, 0, 0);
#else
    *(uint4*)ldst = *(const uint4*)gsrc;
#endif
}

__device__ __forceinline__ void stage_wait() {
#if HAVE_ASYNC_LDS
    __builtin_amdgcn_s_wait_asynccnt(0);
#endif
}

// ---------------------------------------------------------------------------
// Kernel 0: f32 -> bf16 conversion of query, keys (input_flatten), out_w
// ---------------------------------------------------------------------------
__global__ void convert_kernel(const float* __restrict__ q,
                               const float* __restrict__ f,
                               const float* __restrict__ w,
                               u16* __restrict__ qb,
                               u16* __restrict__ fb,
                               u16* __restrict__ wb) {
    int i = blockIdx.x * blockDim.x + threadIdx.x;
    int stride = gridDim.x * blockDim.x;
    for (int t = i; t < LQ * DMODEL; t += stride)  qb[t] = f2bf(q[t]);
    for (int t = i; t < LIN * DMODEL; t += stride) fb[t] = f2bf(f[t]);
    for (int t = i; t < DMODEL * DMODEL; t += stride) wb[t] = f2bf(w[t]);
}

// ---------------------------------------------------------------------------
// Kernel 1: fused scores-GEMM (WMMA bf16) + per-level top-4.
// 128 WGs x 256 thr (8 waves). WG handles 16 queries over all 16384 keys.
// Q tile staged in LDS (async copy), A fragments in registers, B fragments
// double-buffered. Top-4 kept per (wave,lane,row) stream as packed u32 keys:
//   key = (orderable(score) & 0xFFFFF000) | (~idx & 0xFFF)
// (larger key == higher score, lower index on near-tie), merged through LDS.
// ---------------------------------------------------------------------------
__global__ void __launch_bounds__(256, 1)
scores_topk_kernel(const u16* __restrict__ qbf,
                   const u16* __restrict__ fbf,
                   int* __restrict__ topIdx /* [LQ][NLVL][4] */) {
    __shared__ __align__(16) u16 sQ[16 * DMODEL];    // 8 KB
    __shared__ u32 cand[16 * 128 * 4];               // 32 KB
    __shared__ u32 cand2[16 * 16];                   // 1 KB

    const int tid  = threadIdx.x;
    const int lane = tid & 31;
    const int wave = tid >> 5;
    const int n    = lane & 15;
    const int hi   = lane >> 4;
    const int wgq  = blockIdx.x * 16;

    // stage 16x256 bf16 query tile into LDS: 512 x 16B chunks, 2 per thread
    {
        const u16* gsrc = qbf + (size_t)wgq * DMODEL;
        for (int c = tid; c < (16 * DMODEL) / 8; c += 256)
            stage16(gsrc + c * 8, &sQ[c * 8]);
        stage_wait();
    }
    __syncthreads();

    // A fragments from LDS (ds_load_b128): lane row = lane&15, hi selects K+16
    BF16x16 a[8];
    {
        const u16* qrow = &sQ[n * DMODEL];
#pragma unroll
        for (int kk = 0; kk < 8; ++kk) {
            int off = kk * 32 + hi * 16;
            a[kk].u[0] = *(const uint4*)(qrow + off);
            a[kk].u[1] = *(const uint4*)(qrow + off + 8);
        }
    }

    for (int l = 0; l < NLVL; ++l) {
        u32 top[8][4];
#pragma unroll
        for (int r = 0; r < 8; ++r)
#pragma unroll
            for (int e = 0; e < 4; ++e) top[r][e] = 0u;

        for (int it = 0; it < HWL / 128; ++it) {           // 32 iterations
            const int keyLocal = it * 128 + wave * 16 + n; // key within level
            const u16* frow = fbf + (size_t)(l * HWL + keyLocal) * DMODEL;
            if (it + 1 < HWL / 128)
                __builtin_prefetch(frow + 128 * DMODEL, 0, 3);

            // B double buffer: wmma kk only depends on loads issued step kk-1
            BF16x16 b[2];
            b[0].u[0] = *(const uint4*)(frow + hi * 16);
            b[0].u[1] = *(const uint4*)(frow + hi * 16 + 8);
            v8f c = {};
#pragma unroll
            for (int kk = 0; kk < 8; ++kk) {
                const int cur = kk & 1;
                if (kk < 7) {
                    int off = (kk + 1) * 32 + hi * 16;
                    b[cur ^ 1].u[0] = *(const uint4*)(frow + off);
                    b[cur ^ 1].u[1] = *(const uint4*)(frow + off + 8);
                }
                c = __builtin_amdgcn_wmma_f32_16x16x32_bf16(
                        false, a[kk].v, false, b[cur].v, (short)0, c, false, false);
            }

            const u32 nlow = (~(u32)keyLocal) & 0xFFFu;
#pragma unroll
            for (int r = 0; r < 8; ++r) {
                u32 key = (orderable(c[r]) & 0xFFFFF000u) | nlow;
                ins4(key, top[r][0], top[r][1], top[r][2], top[r][3]);
            }
        }

        // dump per-stream top4 to LDS: query q = r + hi*8, stream s = wave*16+n
        const int s = wave * 16 + n;
#pragma unroll
        for (int r = 0; r < 8; ++r) {
            int q = r + hi * 8;
            u32* dst = cand + (size_t)(q * 128 + s) * 4;
            dst[0] = top[r][0]; dst[1] = top[r][1];
            dst[2] = top[r][2]; dst[3] = top[r][3];
        }
        __syncthreads();

        if (tid < 64) {                                   // 16 q x 4 parts
            int q = tid >> 2, part = tid & 3;
            const u32* src = cand + (size_t)(q * 128 + part * 32) * 4;
            u32 t0 = 0, t1 = 0, t2 = 0, t3 = 0;
            for (int i2 = 0; i2 < 128; ++i2) ins4(src[i2], t0, t1, t2, t3);
            u32* dst = cand2 + (size_t)(q * 4 + part) * 4;
            dst[0] = t0; dst[1] = t1; dst[2] = t2; dst[3] = t3;
        }
        __syncthreads();

        if (tid < 16) {
            int q = tid;
            const u32* src = cand2 + (size_t)q * 16;
            u32 t0 = 0, t1 = 0, t2 = 0, t3 = 0;
            for (int i2 = 0; i2 < 16; ++i2) ins4(src[i2], t0, t1, t2, t3);
            int* dst = topIdx + ((size_t)(wgq + q) * NLVL + l) * 4;
            dst[0] = (int)((~t0) & 0xFFFu);
            dst[1] = (int)((~t1) & 0xFFFu);
            dst[2] = (int)((~t2) & 0xFFFu);
            dst[3] = (int)((~t3) & 0xFFFu);
        }
        __syncthreads();
    }
}

// ---------------------------------------------------------------------------
// Kernel 2: neighbor expansion + degenerate bilinear (0.25 x 4-corner gather),
// plus sampling_locations output. One block per query, one thread per channel.
// ---------------------------------------------------------------------------
__global__ void sample_kernel(const float* __restrict__ vflat,
                              const int* __restrict__ topIdx,
                              const int* __restrict__ lsi,
                              float* __restrict__ loc_out /* [LQ][NLVL][36][2] */,
                              u16* __restrict__ headbf /* [LQ][DMODEL] */) {
    const int q = blockIdx.x;
    const int d = threadIdx.x;
    __shared__ int sIdx[NLVL * 4];
    __shared__ int sLsi[NLVL];
    if (d < NLVL * 4) sIdx[d] = topIdx[q * NLVL * 4 + d];
    if (d < NLVL) sLsi[d] = lsi[d];
    __syncthreads();

    float acc = 0.f;
#pragma unroll
    for (int l = 0; l < NLVL; ++l) {
        const float* vbase = vflat + (size_t)sLsi[l] * DMODEL;
#pragma unroll
        for (int p = 0; p < 4; ++p) {
            int idx = sIdx[l * 4 + p];
#pragma unroll
            for (int j = 0; j < 9; ++j) {
                int nb = (j % 3 - 1) + (j / 3 - 1) * GRID_H;
                int r = idx + nb;
                r = min(max(r, 0), CLIP_MAX);
                int ow = r >> 6;
                int oh = r & 63;
                int g = oh * GRID_H + ow;        // gather uses transposed index
                float s = vbase[(size_t)g * DMODEL + d];             // (y1,x1)
                if (ow > 0)           s += vbase[(size_t)(g - 1) * DMODEL + d];
                if (oh > 0)           s += vbase[(size_t)(g - 64) * DMODEL + d];
                if (ow > 0 && oh > 0) s += vbase[(size_t)(g - 65) * DMODEL + d];
                acc += s;
            }
        }
    }
    acc *= 0.25f / 144.f;  // bilinear 0.25 * aw (1/(NLVL*36))
    headbf[(size_t)q * DMODEL + d] = f2bf(acc);

    // sampling_locations: 4 lvl * 36 pts * 2 = 288 values per query
    for (int v = d; v < NLVL * 36 * 2; v += blockDim.x) {
        int l = v / 72;
        int rem = v - l * 72;
        int p36 = rem >> 1;
        int c = rem & 1;
        int j = p36 >> 2;       // neighbor-major flatten (9,4)
        int p = p36 & 3;
        int nb = (j % 3 - 1) + (j / 3 - 1) * GRID_H;
        int r = sIdx[l * 4 + p] + nb;
        r = min(max(r, 0), CLIP_MAX);
        int ow = r >> 6;
        int oh = r & 63;
        loc_out[(size_t)q * 288 + v] = (float)(c == 0 ? ow : oh) * (1.f / 64.f);
    }
}

// ---------------------------------------------------------------------------
// Kernel 3: output projection out = head @ W^T + b via WMMA bf16.
// 128 WGs x 8 waves; A tile staged through LDS, each wave covers two
// 16-column tiles of the 256 outputs.
// ---------------------------------------------------------------------------
__global__ void __launch_bounds__(256, 1)
proj_kernel(const u16* __restrict__ headbf,
            const u16* __restrict__ wbf,
            const float* __restrict__ bias,
            float* __restrict__ out) {
    __shared__ __align__(16) u16 sA[16 * DMODEL];    // 8 KB
    const int tid  = threadIdx.x;
    const int lane = tid & 31;
    const int wave = tid >> 5;
    const int n    = lane & 15;
    const int hi   = lane >> 4;
    const int wgq  = blockIdx.x * 16;

    {
        const u16* gsrc = headbf + (size_t)wgq * DMODEL;
        for (int c = tid; c < (16 * DMODEL) / 8; c += 256)
            stage16(gsrc + c * 8, &sA[c * 8]);
        stage_wait();
    }
    __syncthreads();

    BF16x16 a[8];
    {
        const u16* arow = &sA[n * DMODEL];
#pragma unroll
        for (int kk = 0; kk < 8; ++kk) {
            int off = kk * 32 + hi * 16;
            a[kk].u[0] = *(const uint4*)(arow + off);
            a[kk].u[1] = *(const uint4*)(arow + off + 8);
        }
    }

#pragma unroll
    for (int nt = 0; nt < 2; ++nt) {
        int nBase = (wave * 2 + nt) * 16;
        const u16* wrow = wbf + (size_t)(nBase + n) * DMODEL;  // B col = W row
        BF16x16 b[2];
        b[0].u[0] = *(const uint4*)(wrow + hi * 16);
        b[0].u[1] = *(const uint4*)(wrow + hi * 16 + 8);
        v8f c = {};
#pragma unroll
        for (int kk = 0; kk < 8; ++kk) {
            const int cur = kk & 1;
            if (kk < 7) {
                int off = (kk + 1) * 32 + hi * 16;
                b[cur ^ 1].u[0] = *(const uint4*)(wrow + off);
                b[cur ^ 1].u[1] = *(const uint4*)(wrow + off + 8);
            }
            c = __builtin_amdgcn_wmma_f32_16x16x32_bf16(
                    false, a[kk].v, false, b[cur].v, (short)0, c, false, false);
        }
        int col = nBase + n;
        float bc = bias[col];
#pragma unroll
        for (int r = 0; r < 8; ++r) {
            int m = r + hi * 8;
            out[(size_t)(wgq + m) * DMODEL + col] = c[r] + bc;
        }
    }
}

// ---------------------------------------------------------------------------
extern "C" void kernel_launch(void* const* d_in, const int* in_sizes, int n_in,
                              void* d_out, int out_size, void* d_ws, size_t ws_size,
                              hipStream_t stream) {
    const float* query         = (const float*)d_in[0];
    // d_in[1] reference_points: unused by the reference computation
    const float* input_flatten = (const float*)d_in[2];
    // d_in[3] input_spatial_shapes: fixed 64x64 per level
    const int*   lsi           = (const int*)d_in[4];
    const float* out_w         = (const float*)d_in[5];
    const float* out_b         = (const float*)d_in[6];

    char* ws = (char*)d_ws;
    u16* qbf    = (u16*)(ws);                                  // 1 MB
    u16* fbf    = (u16*)(ws + (1u << 20));                     // 8 MB
    u16* wbf    = (u16*)(ws + (9u << 20));                     // 128 KB
    int* topIdx = (int*)(ws + (9u << 20) + (1u << 17));        // 128 KB
    u16* headbf = (u16*)(ws + (9u << 20) + (2u << 17));        // 1 MB

    float* out_main = (float*)d_out;                 // [2048][256]
    float* loc_out  = out_main + LQ * DMODEL;        // [2048][4][36][2]

    convert_kernel<<<4096, 256, 0, stream>>>(query, input_flatten, out_w,
                                             qbf, fbf, wbf);

    scores_topk_kernel<<<LQ / 16, 256, 0, stream>>>(qbf, fbf, topIdx);

    sample_kernel<<<LQ, 256, 0, stream>>>(input_flatten, topIdx, lsi,
                                          loc_out, headbf);

    proj_kernel<<<LQ / 16, 256, 0, stream>>>(headbf, wbf, out_b, out_main);
}
